// VariationalSDEModel_39281770889750
// MI455X (gfx1250) — matrix operand order repaired and checked by reading, not compiled, not run on hardware
//
#include <hip/hip_runtime.h>
#include <math.h>

#define N_PART 65536
#define T_STEPS 50

#if __has_builtin(__builtin_amdgcn_tanhf)
#warning CDNA5_PROBE: __builtin_amdgcn_tanhf AVAILABLE
#else
#warning CDNA5_PROBE: __builtin_amdgcn_tanhf MISSING (using exp-based fallback)
#endif
#if __has_builtin(__builtin_amdgcn_cvt_pk_bf16_f32)
#warning CDNA5_PROBE: __builtin_amdgcn_cvt_pk_bf16_f32 AVAILABLE
#else
#warning CDNA5_PROBE: __builtin_amdgcn_cvt_pk_bf16_f32 MISSING (using v_perm pack)
#endif

typedef __attribute__((ext_vector_type(16))) __bf16 v16bf;
typedef __attribute__((ext_vector_type(8)))  float  v8f;

struct __align__(16) U4 { unsigned x, y, z, w; };
union __align__(16) AFrag { unsigned u[8]; __bf16 h[16]; };  // one WMMA A/B fragment

__device__ __forceinline__ v16bf asv(const AFrag& f) { return __builtin_bit_cast(v16bf, f); }

// ---------- bf16 helpers ----------
// RNE scalar (cold paths: prep + staging)
__device__ __forceinline__ unsigned short f2bf(float f) {
  unsigned u = __float_as_uint(f);
  u += 0x7FFFu + ((u >> 16) & 1u);
  return (unsigned short)(u >> 16);
}
// hot-path pack: round-half-up + single v_perm_b32 (3 VALU per pair)
#if __has_builtin(__builtin_amdgcn_cvt_pk_bf16_f32)
__device__ __forceinline__ unsigned pk2(float a, float b) {
  return __builtin_bit_cast(unsigned, __builtin_amdgcn_cvt_pk_bf16_f32(a, b));
}
#else
__device__ __forceinline__ unsigned pk2(float a, float b) {
  return __builtin_amdgcn_perm(__float_as_uint(b) + 0x8000u,
                               __float_as_uint(a) + 0x8000u, 0x07060302u);
}
#endif

__device__ __forceinline__ float tanh_f(float x) {
#if __has_builtin(__builtin_amdgcn_tanhf)
  return __builtin_amdgcn_tanhf(x);
#else
  x = fminf(fmaxf(x, -9.0f), 9.0f);
  float e = __expf(2.0f * x);
  return (e - 1.0f) * __builtin_amdgcn_rcpf(e + 1.0f);
#endif
}

// convert 8 consecutive f32 -> 4 packed bf16 pairs
__device__ __forceinline__ U4 cvt8(const float* p) {
  float4 a = *(const float4*)p;
  float4 b = *(const float4*)(p + 4);
  U4 r;
  r.x = pk2(a.x, a.y); r.y = pk2(a.z, a.w);
  r.z = pk2(b.x, b.y); r.w = pk2(b.z, b.w);
  return r;
}

// load two 16B chunks into a fragment
__device__ __forceinline__ void ld2q(AFrag& f, const unsigned short* r) {
  *(U4*)&f.u[0] = *(const U4*)r;
  *(U4*)&f.u[4] = *(const U4*)(r + 16);
}
__device__ __forceinline__ void ld2q32(AFrag& f, const unsigned* r) {
  *(U4*)&f.u[0] = *(const U4*)r;
  *(U4*)&f.u[4] = *(const U4*)(r + 4);
}
__device__ __forceinline__ void ld8f(float* d, const float* r) {
  *(U4*)d = *(const U4*)r;
  *(U4*)(d + 4) = *(const U4*)(r + 4);
}

// ---------- fragment loaders (ISA 7.12.2 layouts, wave32) ----------
__device__ __forceinline__ AFrag ldA_f32(const float* A, int lane, int ch, int lda) {
  int m = lane & 15, ho = (lane & 16) ? 8 : 0;
  const float* r = A + m * lda + ch * 32 + ho;
  AFrag f;
  *(U4*)&f.u[0] = cvt8(r);
  *(U4*)&f.u[4] = cvt8(r + 16);
  return f;
}
__device__ __forceinline__ v16bf ldB(const unsigned short* Bt, int lane, int nt, int ch) {
  int n = nt * 16 + (lane & 15);
  int k = ch * 32 + ((lane & 16) ? 8 : 0);
  AFrag f;
  const U4* p = (const U4*)(Bt + n * 136 + k);
  *(U4*)&f.u[0] = p[0];
  *(U4*)&f.u[4] = p[2];
  return asv(f);
}

// [16 x 128] += [16 x 128] @ [128 x 128]  (A-stationary: 4 K-chunks x 8 N-tiles)
__device__ __forceinline__ void gemm128(v8f acc[8], const AFrag a[4],
                                        const unsigned short* Bt, int lane) {
#pragma unroll
  for (int ch = 0; ch < 4; ++ch) {
#pragma unroll
    for (int nt = 0; nt < 8; ++nt) {
      v16bf b = ldB(Bt, lane, nt, ch);
      acc[nt] = __builtin_amdgcn_wmma_f32_16x16x32_bf16(
          false, asv(a[ch]), false, b, (short)0, acc[nt], false, false);
    }
  }
}

// layer-1 slot pair: act(P + SV + z0*Wz0 + z1*Wz1); __bf16 casts encourage v_fma_mix_f32_bf16
__device__ __forceinline__ unsigned relu_pair(const AFrag& P, const AFrag& S, const AFrag& W0,
                                              const AFrag& W1f, int i, float za, float zb) {
  float a = (float)P.h[2 * i] + (float)S.h[2 * i] + za * (float)W0.h[2 * i] + zb * (float)W1f.h[2 * i];
  float b = (float)P.h[2 * i + 1] + (float)S.h[2 * i + 1] + za * (float)W0.h[2 * i + 1] + zb * (float)W1f.h[2 * i + 1];
  return pk2(fmaxf(a, 0.f), fmaxf(b, 0.f));
}
__device__ __forceinline__ unsigned tanh_pair(const AFrag& P, const AFrag& S, const AFrag& W0,
                                              const AFrag& W1f, int i, float za, float zb) {
  float a = (float)P.h[2 * i] + (float)S.h[2 * i] + za * (float)W0.h[2 * i] + zb * (float)W1f.h[2 * i];
  float b = (float)P.h[2 * i + 1] + (float)S.h[2 * i + 1] + za * (float)W0.h[2 * i + 1] + zb * (float)W1f.h[2 * i + 1];
  return pk2(tanh_f(a), tanh_f(b));
}

// ============================================================================
// Prep: transposed bf16 weight tiles + per-step layer-1 constant vectors
// ws layout (unsigned short elems): [0]=W1ctxT 16384, [16384]=V1ctxT,
// [32768]=W2T, [49152]=V2T, [65536]=SVd 50*128, [71936]=SVc 50*128
// ============================================================================
__global__ void sde_prep_kernel(const float* __restrict__ times, const float* __restrict__ freqs,
                                const float* __restrict__ W1, const float* __restrict__ b1,
                                const float* __restrict__ W2, const float* __restrict__ V1,
                                const float* __restrict__ c1, const float* __restrict__ V2,
                                unsigned short* __restrict__ ws) {
  int bid = blockIdx.x, tid = threadIdx.x;
  if (bid < 64) {
    int i = bid * 256 + tid;          // 0..16383
    int o = i >> 7, c = i & 127;      // o = output unit, c = input index
    ws[0     + o * 128 + c] = f2bf(W1[(2 + c) * 128 + o]);  // W1ctxT[n][k]
    ws[16384 + o * 128 + c] = f2bf(V1[(2 + c) * 128 + o]);  // V1ctxT
    ws[32768 + o * 128 + c] = f2bf(W2[c * 128 + o]);        // W2T[n][k]
    ws[49152 + o * 128 + c] = f2bf(V2[c * 128 + o]);        // V2T
  } else {
    int s = bid - 64;                 // 0..49
    if (s < T_STEPS && tid < 128) {
      float t = times[s];
      float acc = b1[tid];
      for (int j = 0; j < 16; ++j) {
        float arg = 6.283185307179586f * t * freqs[j];
        acc += sinf(arg) * W1[(130 + j) * 128 + tid];
        acc += cosf(arg) * W1[(146 + j) * 128 + tid];
      }
      ws[65536 + s * 128 + tid] = f2bf(acc);
      ws[71936 + s * 128 + tid] = f2bf(c1[tid] + t * V1[130 * 128 + tid]);
    }
  }
}

// ============================================================================
// Persistent SDE kernel: 8 waves/WG, 16 particles/wave, all 50 steps in-kernel.
// All loop-invariant per-lane data (P/Q projections, swizzled z-weights,
// layer-3 columns) lives in LDS so the kernel stays under 256 VGPRs.
// ============================================================================
__global__ __launch_bounds__(256)
void sde_main_kernel(const float* __restrict__ z0p, const float* __restrict__ pctx,
                     const float* __restrict__ cctx, const float* __restrict__ times,
                     const float* __restrict__ xi, const float* __restrict__ log_diff,
                     const float* __restrict__ W1, const float* __restrict__ b2,
                     const float* __restrict__ W3, const float* __restrict__ b3,
                     const float* __restrict__ V1, const float* __restrict__ c2,
                     const float* __restrict__ V3, const float* __restrict__ c3,
                     const unsigned short* __restrict__ ws, float* __restrict__ outp) {
  __shared__ unsigned short sA[128 * 136];  // W1ctxT -> per-wave P staging (persists)
  __shared__ unsigned short sB[128 * 136];  // V1ctxT -> per-wave Q staging (persists)
  __shared__ unsigned short sC[128 * 136];  // W2T (resident)
  __shared__ unsigned short sD[128 * 136];  // V2T (resident)
  __shared__ unsigned sLN[32 * 180];        // per-lane: [0..127]=z-weights, [128..175]=layer-3 consts
  __shared__ float sDX[8][16][2];           // per-wave drift exchange

  const int tid = threadIdx.x;
  const int lane = tid & 31, wave = tid >> 5;
  const int nl = lane & 15, hsel = (lane >> 4) & 1, ho = hsel * 8;
  const int rowbase = blockIdx.x * 128 + wave * 16;
  const int grow = rowbase + nl;

  // cooperative load of all four weight tiles (u32 granularity, pad stride 136)
  for (int i = tid; i < 8192; i += 256) {
    int row = i >> 6, col = i & 63;
    ((unsigned*)(sA + row * 136))[col] = ((const unsigned*)(ws + 0     + row * 128))[col];
    ((unsigned*)(sB + row * 136))[col] = ((const unsigned*)(ws + 16384 + row * 128))[col];
    ((unsigned*)(sC + row * 136))[col] = ((const unsigned*)(ws + 32768 + row * 128))[col];
    ((unsigned*)(sD + row * 136))[col] = ((const unsigned*)(ws + 49152 + row * 128))[col];
  }
  // wave 0 builds the per-lane table: swizzled layer-1 z-weights + layer-3 columns
  if (tid < 32) {
    int l = tid, lho = (l & 16) ? 8 : 0;
    unsigned* wz = sLN + l * 180;
#pragma unroll
    for (int ch = 0; ch < 4; ++ch) {
#pragma unroll
      for (int i = 0; i < 8; ++i) {
        int k = ch * 32 + lho + ((i < 4) ? 2 * i : 16 + 2 * (i - 4));
        float2 a0 = *(const float2*)(W1 + k);
        float2 a1 = *(const float2*)(W1 + 128 + k);
        float2 b0 = *(const float2*)(V1 + k);
        float2 b1v = *(const float2*)(V1 + 128 + k);
        wz[0 * 32 + ch * 8 + i] = pk2(a0.x, a0.y);
        wz[1 * 32 + ch * 8 + i] = pk2(a1.x, a1.y);
        wz[2 * 32 + ch * 8 + i] = pk2(b0.x, b0.y);
        wz[3 * 32 + ch * 8 + i] = pk2(b1v.x, b1v.y);
      }
    }
    float* lf = (float*)wz;
    int n0 = l & 15;
#pragma unroll
    for (int nt = 0; nt < 8; ++nt) {
      int n = nt * 16 + n0;
      lf[128 + nt] = W3[n * 2];
      lf[136 + nt] = W3[n * 2 + 1];
      lf[144 + nt] = b2[n];
      lf[152 + nt] = V3[n * 2];
      lf[160 + nt] = V3[n * 2 + 1];
      lf[168 + nt] = c2[n];
    }
  }
  __syncthreads();

  v8f zz = {};
  {
    AFrag a[4];
    v8f acc[8];
    // ---- P = posterior_ctx @ W1ctx ----
#pragma unroll
    for (int nt = 0; nt < 8; ++nt) acc[nt] = zz;
#pragma unroll
    for (int ch = 0; ch < 4; ++ch) a[ch] = ldA_f32(pctx + (size_t)rowbase * 128, lane, ch, 128);
    gemm128(acc, a, sA, lane);
    __syncthreads();                  // everyone done reading W1ctxT; reuse sA as P staging
    {
      unsigned short* stg = sA + wave * 16 * 136;
#pragma unroll
      for (int nt = 0; nt < 8; ++nt)
#pragma unroll
        for (int j = 0; j < 8; ++j)
          stg[(j + ho) * 136 + nt * 16 + nl] = f2bf(acc[nt][j]);   // C-layout -> row-major
    }
    // ---- Q = cnf_ctx @ V1ctx ----
#pragma unroll
    for (int nt = 0; nt < 8; ++nt) acc[nt] = zz;
#pragma unroll
    for (int ch = 0; ch < 4; ++ch) a[ch] = ldA_f32(cctx + (size_t)rowbase * 128, lane, ch, 128);
    gemm128(acc, a, sB, lane);
    __syncthreads();                  // everyone done reading V1ctxT; reuse sB as Q staging
    {
      unsigned short* stg = sB + wave * 16 * 136;
#pragma unroll
      for (int nt = 0; nt < 8; ++nt)
#pragma unroll
        for (int j = 0; j < 8; ++j)
          stg[(j + ho) * 136 + nt * 16 + nl] = f2bf(acc[nt][j]);
    }
    asm volatile("s_wait_dscnt 0" ::: "memory");
  }

  float db0 = b3[0] + c3[0], db1 = b3[1] + c3[1];
  float sp = log1pf(__expf(log_diff[0]));   // softplus(log_diff)

  float2 zi = *(const float2*)(z0p + (size_t)grow * 2);
  float za = zi.x, zb = zi.y;
  if (!hsel) *(float2*)(outp + (size_t)grow * 2) = zi;   // traj[0] = z0

  const unsigned short* SVdAll = ws + 65536;
  const unsigned short* SVcAll = ws + 71936;
  const unsigned short* pRow = sA + (wave * 16 + nl) * 136 + ho;  // this lane's P row slice
  const unsigned short* qRow = sB + (wave * 16 + nl) * 136 + ho;  // this lane's Q row slice
  const unsigned* wz = sLN + lane * 180;
  const float* lcf = (const float*)wz;

  float tcur = times[0];
  for (int s = 0; s < T_STEPS; ++s) {
    float tnext = times[s + 1];
    float dt = tnext - tcur;
    float sig = sp * (1.0f - tcur) * sqrtf(fmaxf(dt, 1e-12f));
    tcur = tnext;
    float2 x = *(const float2*)(xi + ((size_t)s * N_PART + grow) * 2);  // hide under GEMMs
    if (s + 1 < T_STEPS)
      __builtin_prefetch((const void*)(xi + ((size_t)(s + 1) * N_PART + grow) * 2), 0, 1);

    AFrag a[4];
    v8f acc[8];
    float p0[8], p1[8];
#pragma unroll
    for (int j = 0; j < 8; ++j) { p0[j] = 0.f; p1[j] = 0.f; }

    // -------- drift net --------
    const unsigned short* svd = SVdAll + s * 128;
#pragma unroll
    for (int ch = 0; ch < 4; ++ch) {
      AFrag P, S, W0, W1f;
      ld2q(P, pRow + ch * 32);
      ld2q(S, svd + ch * 32 + ho);
      ld2q32(W0, wz + 0 * 32 + ch * 8);
      ld2q32(W1f, wz + 1 * 32 + ch * 8);
#pragma unroll
      for (int i = 0; i < 8; ++i)
        a[ch].u[i] = relu_pair(P, S, W0, W1f, i, za, zb);
    }
#pragma unroll
    for (int nt = 0; nt < 8; ++nt) acc[nt] = zz;
    gemm128(acc, a, sC, lane);
    {
      float w3a[8], w3b[8], bb2[8];
      ld8f(w3a, lcf + 128); ld8f(w3b, lcf + 136); ld8f(bb2, lcf + 144);
#pragma unroll
      for (int nt = 0; nt < 8; ++nt)
#pragma unroll
        for (int j = 0; j < 8; ++j) {
          float v = fmaxf(acc[nt][j] + bb2[nt], 0.f);
          p0[j] += v * w3a[nt];
          p1[j] += v * w3b[nt];
        }
    }

    // -------- cnf field net --------
    const unsigned short* svc = SVcAll + s * 128;
#pragma unroll
    for (int ch = 0; ch < 4; ++ch) {
      AFrag P, S, W0, W1f;
      ld2q(P, qRow + ch * 32);
      ld2q(S, svc + ch * 32 + ho);
      ld2q32(W0, wz + 2 * 32 + ch * 8);
      ld2q32(W1f, wz + 3 * 32 + ch * 8);
#pragma unroll
      for (int i = 0; i < 8; ++i)
        a[ch].u[i] = tanh_pair(P, S, W0, W1f, i, za, zb);
    }
#pragma unroll
    for (int nt = 0; nt < 8; ++nt) acc[nt] = zz;
    gemm128(acc, a, sD, lane);
    {
      float v3a[8], v3b[8], cc2[8];
      ld8f(v3a, lcf + 152); ld8f(v3b, lcf + 160); ld8f(cc2, lcf + 168);
#pragma unroll
      for (int nt = 0; nt < 8; ++nt)
#pragma unroll
        for (int j = 0; j < 8; ++j) {
          float v = tanh_f(acc[nt][j] + cc2[nt]);
          p0[j] += v * v3a[nt];
          p1[j] += v * v3b[nt];
        }
    }

    // reduce over the 16 lanes of each half (xor masks 1..8 stay in-half)
#pragma unroll
    for (int j = 0; j < 8; ++j) {
#pragma unroll
      for (int m = 1; m < 16; m <<= 1) {
        p0[j] += __shfl_xor(p0[j], m, 32);
        p1[j] += __shfl_xor(p1[j], m, 32);
      }
    }
    // exchange row drifts across halves via per-wave LDS slot
#pragma unroll
    for (int j = 0; j < 8; ++j)
      if (nl == j) { sDX[wave][j + ho][0] = p0[j]; sDX[wave][j + ho][1] = p1[j]; }
    asm volatile("s_wait_dscnt 0" ::: "memory");
    float d0 = sDX[wave][nl][0] + db0;
    float d1 = sDX[wave][nl][1] + db1;

    za += d0 * dt + x.x * sig;
    zb += d1 * dt + x.y * sig;
    if (!hsel)
      *(float2*)(outp + ((size_t)(s + 1) * N_PART + grow) * 2) = make_float2(za, zb);
  }
}

extern "C" void kernel_launch(void* const* d_in, const int* in_sizes, int n_in,
                              void* d_out, int out_size, void* d_ws, size_t ws_size,
                              hipStream_t stream) {
  const float* z0p   = (const float*)d_in[0];
  const float* pctx  = (const float*)d_in[1];
  const float* cctx  = (const float*)d_in[2];
  const float* times = (const float*)d_in[3];
  const float* xi    = (const float*)d_in[4];
  const float* freqs = (const float*)d_in[5];
  const float* ldif  = (const float*)d_in[6];
  const float* W1    = (const float*)d_in[7];
  const float* b1    = (const float*)d_in[8];
  const float* W2    = (const float*)d_in[9];
  const float* b2    = (const float*)d_in[10];
  const float* W3    = (const float*)d_in[11];
  const float* b3    = (const float*)d_in[12];
  const float* V1    = (const float*)d_in[13];
  const float* c1    = (const float*)d_in[14];
  const float* V2    = (const float*)d_in[15];
  const float* c2    = (const float*)d_in[16];
  const float* V3    = (const float*)d_in[17];
  const float* c3    = (const float*)d_in[18];
  unsigned short* ws = (unsigned short*)d_ws;
  float* outp = (float*)d_out;

  sde_prep_kernel<<<114, 256, 0, stream>>>(times, freqs, W1, b1, W2, V1, c1, V2, ws);
  sde_main_kernel<<<N_PART / 128, 256, 0, stream>>>(z0p, pctx, cctx, times, xi, ldif,
                                                    W1, b2, W3, b3, V1, c2, V3, c3,
                                                    ws, outp);
}
